// GatherConvND_29583734735363
// MI455X (gfx1250) — matrix operand order repaired and checked by reading, not compiled, not run on hardware
//
#include <hip/hip_runtime.h>

// ---------------- problem constants (fixed hyperparameters) ----------------
#define C_DIM   512
#define H_DIM   8
#define D_DIM   64
#define K_TAB   64
#define S_TAPS  33
#define L_SEQ   4096
#define MAXFREQ 16.0f
#define MINFREQ 1.0f
#define MAXRECP 256.0f   // HALF_S * MAX_FREQ

// LDS tile row stride (shorts): 32 data + 8 pad -> 80B rows, 16B aligned,
// fragment reads from lanes 0-15 land on 16 distinct banks.
#define TSTRIDE 40

typedef __attribute__((ext_vector_type(16))) __bf16 v16bf;
typedef __attribute__((ext_vector_type(8)))  float  v8f;
typedef __attribute__((__vector_size__(4 * sizeof(int)))) int v4i;

// Async global->LDS path (CDNA5 GLOBAL_LOAD_ASYNC_TO_LDS_B128, ASYNCcnt).
#ifndef USE_ASYNC_LDS
#if defined(__has_builtin)
#if __has_builtin(__builtin_amdgcn_global_load_async_to_lds_b128)
#define USE_ASYNC_LDS 1
#endif
#endif
#endif
#ifndef USE_ASYNC_LDS
#define USE_ASYNC_LDS 0
#endif

#if USE_ASYNC_LDS
typedef __attribute__((address_space(1))) v4i GV4;   // global int4
typedef __attribute__((address_space(3))) v4i LV4;   // LDS int4
__device__ __forceinline__ void async_copy16(const void* g, void* l) {
  __builtin_amdgcn_global_load_async_to_lds_b128((GV4*)g, (LV4*)l, 0, 0);
}
__device__ __forceinline__ void async_wait0() {
  asm volatile("s_wait_asynccnt 0x0" ::: "memory");
}
#endif

__device__ __forceinline__ unsigned short f2bf(float f) {
  union { float f; unsigned u; } v; v.f = f;
  unsigned u = v.u;
  unsigned r = 0x7FFFu + ((u >> 16) & 1u);   // round-to-nearest-even
  return (unsigned short)((u + r) >> 16);
}

__device__ __forceinline__ float silu_f(float v) { return v / (1.f + __expf(-v)); }
__device__ __forceinline__ float sigm_f(float v) { return 1.f / (1.f + __expf(-v)); }

// ---------------- fp32 -> bf16 conversion (row-major) ----------------
__global__ void cvt_bf16_kernel(const float* __restrict__ src,
                                unsigned short* __restrict__ dst, int n) {
  int i = blockIdx.x * blockDim.x + threadIdx.x;
  if (i < n) dst[i] = f2bf(src[i]);
}

// ------- fp32 [K][N] -> bf16 transposed [N][K], 32x32 LDS-tiled -------
__global__ void cvt_bf16_t_kernel(const float* __restrict__ src,
                                  unsigned short* __restrict__ dst,
                                  int K, int N) {
  __shared__ float tile[32][33];
  const int bx = blockIdx.x * 32;          // n block
  const int by = blockIdx.y * 32;          // k block
  const int tx = threadIdx.x & 31;
  const int ty = threadIdx.x >> 5;         // 0..7
  #pragma unroll
  for (int i = 0; i < 32; i += 8)
    tile[ty + i][tx] = src[(size_t)(by + ty + i) * N + bx + tx];
  __syncthreads();
  #pragma unroll
  for (int i = 0; i < 32; i += 8)
    dst[(size_t)(bx + ty + i) * K + by + tx] = f2bf(tile[tx][ty + i]);
}

// ---------------- wave params: one wave32 per row ----------------
__global__ void wave_kernel(const float* __restrict__ x,
                            const float* __restrict__ Ww,
                            const float* __restrict__ bw,
                            float* __restrict__ freq_avg,
                            float* __restrict__ phase_avg, int Mrows) {
  const int wid  = threadIdx.x >> 5;
  const int lane = threadIdx.x & 31;
  const int row  = blockIdx.x * 8 + wid;
  if (row >= Mrows) return;
  const float* xr = x + (size_t)row * C_DIM;

  float acc[16];
  #pragma unroll
  for (int j = 0; j < 16; ++j) acc[j] = 0.f;

  #pragma unroll 4
  for (int i = 0; i < C_DIM / 32; ++i) {
    int c = lane + i * 32;
    float xv = xr[c];
    const float* wr = Ww + (size_t)c * 16;
    #pragma unroll
    for (int j = 0; j < 16; ++j) acc[j] += xv * wr[j];
  }
  #pragma unroll
  for (int j = 0; j < 16; ++j)
    #pragma unroll
    for (int off = 16; off > 0; off >>= 1)
      acc[j] += __shfl_xor(acc[j], off, 32);

  if (lane == 0) {
    float fsum = 0.f, psum = 0.f;
    #pragma unroll
    for (int h = 0; h < H_DIM; ++h) {
      float wf = silu_f(acc[h] + bw[h]);
      float wp = silu_f(acc[8 + h] + bw[8 + h]);
      fsum += sigm_f(wf) * (MAXFREQ - MINFREQ) + MINFREQ;
      psum += tanhf(wp) * MAXFREQ;
    }
    freq_avg[row]  = fsum * (1.f / H_DIM);
    phase_avg[row] = psum * (1.f / H_DIM);
  }
}

// ---------------- bf16 WMMA GEMM: Out = silu(A @ Wt^T + bias) ----------------
// A:  [M,K] bf16 row-major.
// Wt: [N,K] bf16 row-major (weights pre-transposed once in workspace).
// Out:[M,N] fp32.
// Block tile 128x128x32, 256 threads = 8 wave32 (4m x 2n), each wave 32x64 =
// 2x4 tiles of v_wmma_f32_16x16x32_bf16. Both LDS tiles are [row][k] with a
// padded stride; staging is pure per-lane 16B copies (async-to-LDS capable).
__global__ void __launch_bounds__(256, 2)
gemm_bf16_silu_kernel(const unsigned short* __restrict__ A,
                      const unsigned short* __restrict__ Wt,
                      const float* __restrict__ bias,
                      float* __restrict__ Out,
                      int M, int N, int K) {
  __shared__ __align__(16) unsigned short As[2][128 * TSTRIDE];  // [m][k]
  __shared__ __align__(16) unsigned short Bs[2][128 * TSTRIDE];  // [n][k]

  const int t      = threadIdx.x;
  const int lane   = t & 31;
  const int wid    = t >> 5;
  const int wave_m = wid >> 1;   // 0..3
  const int wave_n = wid & 1;    // 0..1
  const int m0     = blockIdx.y * 128;
  const int n0     = blockIdx.x * 128;
  const int nk     = K >> 5;

  // Stage one 128x32 tile of src ([rows][K] row-major) into dstLds.
  auto stage_tile = [&](const unsigned short* src, int row0, int k0,
                        unsigned short* dstLds) {
    #pragma unroll
    for (int i = 0; i < 2; ++i) {
      int c   = t + (i << 8);
      int row = c >> 2;            // 0..127
      int col = (c & 3) << 3;      // 0,8,16,24
      const void* g = src + (size_t)(row0 + row) * K + k0 + col;
      void*       l = &dstLds[row * TSTRIDE + col];
#if USE_ASYNC_LDS
      async_copy16(g, l);
#else
      *reinterpret_cast<uint4*>(l) = *reinterpret_cast<const uint4*>(g);
#endif
    }
  };
  auto load_stage = [&](int kt, int buf) {
    const int k0 = kt << 5;
    stage_tile(A,  m0, k0, As[buf]);
    stage_tile(Wt, n0, k0, Bs[buf]);
  };

  v8f acc[2][4] = {};
  load_stage(0, 0);

  for (int kt = 0; kt < nk; ++kt) {
#if USE_ASYNC_LDS
    async_wait0();                 // stage (kt) fully landed in LDS
#endif
    __syncthreads();
    if (kt + 1 < nk) load_stage(kt + 1, (kt + 1) & 1);

    const unsigned short* as = As[kt & 1];
    const unsigned short* bs = Bs[kt & 1];

    // A fragment (16-bit A 16x32): lanes 0-15 -> M=lane, K 0..7 & 16..23;
    // lanes 16-31 -> M=lane-16, K 8..15 & 24..31. Two 16B LDS reads/tile.
    union FragBF { uint4 u[2]; v16bf v; } af[2], bf[4];
    const int am  = wave_m * 32 + (lane & 15);
    const int akb = (lane >> 4) << 3;
    #pragma unroll
    for (int tm = 0; tm < 2; ++tm) {
      const unsigned short* p = as + (size_t)(am + tm * 16) * TSTRIDE + akb;
      af[tm].u[0] = *reinterpret_cast<const uint4*>(p);
      af[tm].u[1] = *reinterpret_cast<const uint4*>(p + 16);
    }
    // B fragment: column n = lane&15, K (lane>>4)*8 + 0..7 & +16..23, packed 2/VGPR.
    const int bn  = wave_n * 64 + (lane & 15);
    const int bkb = (lane >> 4) << 3;
    #pragma unroll
    for (int tn = 0; tn < 4; ++tn) {
      const unsigned short* p = bs + (size_t)(bn + tn * 16) * TSTRIDE + bkb;
      bf[tn].u[0] = *reinterpret_cast<const uint4*>(p);
      bf[tn].u[1] = *reinterpret_cast<const uint4*>(p + 16);
    }

    #pragma unroll
    for (int tm = 0; tm < 2; ++tm)
      #pragma unroll
      for (int tn = 0; tn < 4; ++tn)
        acc[tm][tn] = __builtin_amdgcn_wmma_f32_16x16x32_bf16(
            false, af[tm].v, false, bf[tn].v, (short)0, acc[tm][tn],
            false, false);
  }

  // Epilogue: C/D layout -> VGPR r: lanes 0-15 M=r, lanes 16-31 M=r+8; N=lane&15
  const int nl = lane & 15;
  const int mh = (lane >> 4) << 3;
  #pragma unroll
  for (int tm = 0; tm < 2; ++tm) {
    #pragma unroll
    for (int tn = 0; tn < 4; ++tn) {
      int gn = n0 + wave_n * 64 + tn * 16 + nl;
      float bi = bias ? bias[gn] : 0.f;
      #pragma unroll
      for (int r = 0; r < 8; ++r) {
        int gm = m0 + wave_m * 32 + tm * 16 + r + mh;
        Out[(size_t)gm * N + gn] = silu_f(acc[tm][tn][r] + bi);
      }
    }
  }
}

// ---------------- gather + kernel interpolation: one block per position ----
__global__ void __launch_bounds__(256)
gather_kernel(const float* __restrict__ x,
              const float* __restrict__ kmax,
              const float* __restrict__ freq_avg,
              const float* __restrict__ phase_avg,
              unsigned short* __restrict__ mix, int L) {
  const int r = blockIdx.x;            // 0 .. B*L-1
  const int b = r / L;
  const int l = r - b * L;
  const int t = threadIdx.x;

  __shared__ float s_kw[H_DIM][S_TAPS];
  __shared__ int   s_idx[S_TAPS];
  __shared__ int   s_ifl[S_TAPS];
  __shared__ float s_wc[S_TAPS];
  __shared__ float s_valid[S_TAPS];

  const float freq  = freq_avg[r];
  const float phase = phase_avg[r];

  if (t < S_TAPS) {
    float rel = (float)(t - 16) * freq + phase;
    float sp  = (float)l + rel;
    bool valid = (sp >= 0.f) && (sp < (float)L);
    int idx = (int)sp;                       // trunc toward zero, like astype(i32)
    idx = idx < 0 ? 0 : (idx > L - 1 ? L - 1 : idx);
    s_idx[t]   = idx;
    s_valid[t] = valid ? 1.f : 0.f;
    float np = (rel + MAXRECP) / (2.f * MAXRECP);
    np = fminf(fmaxf(np, 0.f), 1.f);
    float idxf = np * (float)(K_TAB - 1);
    int ifl = (int)idxf;
    ifl = ifl < 0 ? 0 : (ifl > K_TAB - 2 ? K_TAB - 2 : ifl);
    s_ifl[t] = ifl;
    s_wc[t]  = idxf - (float)ifl;
  }
  __syncthreads();

  if (t < H_DIM * S_TAPS) {
    int h = t / S_TAPS, s = t - h * S_TAPS;
    const float* km = kmax + (size_t)r * C_DIM + h * K_TAB;
    int ifl = s_ifl[s];
    float w = s_wc[s];
    s_kw[h][s] = (km[ifl] * (1.f - w) + km[ifl + 1] * w) * s_valid[s];
  }
  __syncthreads();

  if (t < H_DIM) {
    float sum = 0.f;
    #pragma unroll
    for (int s = 0; s < S_TAPS; ++s) sum += s_kw[t][s];
    float inv = 1.f / (sum + 1e-8f);
    #pragma unroll
    for (int s = 0; s < S_TAPS; ++s) s_kw[t][s] *= inv;
  }
  __syncthreads();

  const float* xb = x + (size_t)b * L * C_DIM;
  #pragma unroll
  for (int i = 0; i < 2; ++i) {
    int c = t + (i << 8);
    int h = c >> 6;                       // c / D_DIM
    float acc = 0.f;
    for (int s = 0; s < S_TAPS; ++s) {
      const float* src = xb + (size_t)s_idx[s] * C_DIM + c;
      if (s + 1 < S_TAPS)
        __builtin_prefetch(xb + (size_t)s_idx[s + 1] * C_DIM + c, 0, 1);
      acc += s_kw[h][s] * (*src);
    }
    mix[(size_t)r * C_DIM + c] = f2bf(acc);
  }
}

// ---------------- host-side launch ----------------
extern "C" void kernel_launch(void* const* d_in, const int* in_sizes, int n_in,
                              void* d_out, int out_size, void* d_ws, size_t ws_size,
                              hipStream_t stream) {
  const float* x     = (const float*)d_in[0];
  const float* Wwave = (const float*)d_in[1];
  const float* bwave = (const float*)d_in[2];
  const float* Wker  = (const float*)d_in[3];
  const float* bker  = (const float*)d_in[4];
  const float* Wout  = (const float*)d_in[5];
  float* out         = (float*)d_out;

  const int BL = in_sizes[0] / C_DIM;     // B * L = 8192

  char* ws = (char*)d_ws;
  size_t off = 0;
  unsigned short* x_bf  = (unsigned short*)(ws + off); off += (size_t)BL * C_DIM * 2;
  unsigned short* wk_t  = (unsigned short*)(ws + off); off += (size_t)C_DIM * C_DIM * 2;
  unsigned short* wo_t  = (unsigned short*)(ws + off); off += (size_t)C_DIM * C_DIM * 2;
  float*          kmax  = (float*)(ws + off);          off += (size_t)BL * C_DIM * 4;
  float*          favg  = (float*)(ws + off);          off += (size_t)BL * 4;
  float*          pavg  = (float*)(ws + off);          off += (size_t)BL * 4;
  unsigned short* mix   = (unsigned short*)(ws + off); off += (size_t)BL * C_DIM * 2;

  // 1) bf16 conversions; weights transposed to [N][K] so GEMM staging is
  //    pure row copies on both operands.
  {
    int n = BL * C_DIM;
    cvt_bf16_kernel<<<(n + 255) / 256, 256, 0, stream>>>(x, x_bf, n);
    dim3 tg(C_DIM / 32, C_DIM / 32);
    cvt_bf16_t_kernel<<<tg, 256, 0, stream>>>(Wker, wk_t, C_DIM, C_DIM);
    cvt_bf16_t_kernel<<<tg, 256, 0, stream>>>(Wout, wo_t, C_DIM, C_DIM);
  }

  // 2) wave params
  wave_kernel<<<BL / 8, 256, 0, stream>>>(x, Wwave, bwave, favg, pavg, BL);

  // 3) kernel_max = silu(x @ W_kernel + b_kernel)   [WMMA GEMM 1]
  gemm_bf16_silu_kernel<<<dim3(C_DIM / 128, BL / 128), 256, 0, stream>>>(
      x_bf, wk_t, bker, kmax, BL, C_DIM, C_DIM);

  // 4) gather + kernel mix -> bf16
  gather_kernel<<<BL, 256, 0, stream>>>(x, kmax, favg, pavg, mix, L_SEQ);

  // 5) out = silu(mix @ W_out)   [WMMA GEMM 2]
  gemm_bf16_silu_kernel<<<dim3(C_DIM / 128, BL / 128), 256, 0, stream>>>(
      mix, wo_t, nullptr, out, BL, C_DIM, C_DIM);
}
